// LSTM_10170482557455
// MI455X (gfx1250) — matrix operand order repaired and checked by reading, not compiled, not run on hardware
//
#include <hip/hip_runtime.h>
#include <stdint.h>

typedef __attribute__((ext_vector_type(16))) _Float16 v16h;
typedef __attribute__((ext_vector_type(8)))  _Float16 v8h;
typedef __attribute__((ext_vector_type(8)))  float    v8f;

#define B_      64
#define T_      512
#define FIN_    264
#define FLSTM_  256
#define H_      512
#define G4_     2048      // 4*H
#define NBLK_   16        // scan workgroups (one per WGP, co-resident)
#define CPB_    32        // hidden columns owned per scan block
#define ZPB_    128       // 4 gates * 32
#define KPAD_   520       // LDS row stride in halves (mult of 8 -> 16B aligned, bank-spread)

#define SCAN_LDS_BYTES ((ZPB_ + B_) * KPAD_ * 2)

// ---------------------------------------------------------------- math helpers
__device__ __forceinline__ float sigf(float x) {
    return 1.0f / (1.0f + __expf(-x));
}
__device__ __forceinline__ float tanhfast(float x) {
    x = fminf(fmaxf(x, -10.0f), 10.0f);
    float e = __expf(2.0f * x);
    return (e - 1.0f) / (e + 1.0f);
}

// ------------------------------------------------- WMMA fragment loads from LDS
// A (16x32 f16, M x K): lane L: m = mbase + L%16; k = kbase + (L/16)*8 + {0..7}
// for regs 0..3 and +16 for regs 4..7 (ISA 7.12.2 16-bit A layout).
__device__ __forceinline__ v16h frag_a_lds(const _Float16* p, int stride,
                                           int mbase, int kbase, int lane) {
    const int half = lane >> 4;
    const _Float16* row = p + (mbase + (lane & 15)) * stride + kbase + half * 8;
    v8h lo = *(const v8h*)(row);
    v8h hi = *(const v8h*)(row + 16);
    v16h a;
#pragma unroll
    for (int i = 0; i < 8; ++i) { a[i] = lo[i]; a[8 + i] = hi[i]; }
    return a;
}
// B (32x16 f16, K x N) stored column-major in LDS as [n][k]:
// lane L: n = cbase + L%16; k = kbase + (L/16)*8 + {0..7}, +16 for regs 4..7.
__device__ __forceinline__ v16h frag_b_lds(const _Float16* p, int stride,
                                           int cbase, int kbase, int lane) {
    const int half = lane >> 4;
    const _Float16* q = p + (cbase + (lane & 15)) * stride + kbase + half * 8;
    v8h lo = *(const v8h*)(q);
    v8h hi = *(const v8h*)(q + 16);
    v16h b;
#pragma unroll
    for (int i = 0; i < 8; ++i) { b[i] = lo[i]; b[8 + i] = hi[i]; }
    return b;
}

// ---------------------------------------------------------------- XW GEMM
// C[R x 2048] (f16) = A[R x K] @ W[K x 2048], f32 accumulate via WMMA.
// Block tile 64 rows x 128 cols, 256 threads = 8 waves, wave -> (mtile, 4 ntiles).
template <bool A_IS_F16>
__global__ void __launch_bounds__(256)
gemm_xw_kernel(const void* __restrict__ Aptr, int lda, int K,
               const float* __restrict__ W, _Float16* __restrict__ C) {
    __shared__ _Float16 aT[64 * 40];    // [m][k], stride 40
    __shared__ _Float16 bT[128 * 40];   // [n][k], stride 40

    const int tid   = threadIdx.x;
    const int lane  = tid & 31;
    const int wave  = tid >> 5;
    const int mtile = wave & 3;
    const int ngrp  = wave >> 2;        // 0..1 -> ntiles ngrp*4 .. +3
    const int rowbase = blockIdx.y * 64;
    const int colbase = blockIdx.x * 128;

    v8f acc[4];
#pragma unroll
    for (int nt = 0; nt < 4; ++nt) acc[nt] = (v8f){0,0,0,0,0,0,0,0};

    for (int kb = 0; kb < K; kb += 32) {
        // stage A tile (64x32) -> f16
        for (int i = tid; i < 64 * 32; i += 256) {
            int m = i >> 5, k = i & 31;
            float v;
            if (A_IS_F16)
                v = (float)((const _Float16*)Aptr)[(long)(rowbase + m) * lda + kb + k];
            else
                v = ((const float*)Aptr)[(long)(rowbase + m) * lda + kb + k];
            aT[m * 40 + k] = (_Float16)v;
        }
        // stage B tile (32x128) transposed -> bT[n][k] (coalesced over n)
        for (int i = tid; i < 128 * 32; i += 256) {
            int n = i & 127, k = i >> 7;
            bT[n * 40 + k] = (_Float16)W[(long)(kb + k) * G4_ + colbase + n];
        }
        __syncthreads();

        v16h a = frag_a_lds(aT, 40, mtile * 16, 0, lane);
#pragma unroll
        for (int nt = 0; nt < 4; ++nt) {
            v16h b = frag_b_lds(bT, 40, (ngrp * 4 + nt) * 16, 0, lane);
            acc[nt] = __builtin_amdgcn_wmma_f32_16x16x32_f16(
                false, a, false, b, (short)0, acc[nt], false, false);
        }
        __syncthreads();
    }

    const int half = lane >> 4, n = lane & 15;
#pragma unroll
    for (int nt = 0; nt < 4; ++nt) {
        int col = colbase + (ngrp * 4 + nt) * 16 + n;
#pragma unroll
        for (int v = 0; v < 8; ++v) {
            int row = rowbase + mtile * 16 + 8 * half + v;
            // streamed exactly once by the scan -> keep it out of L2's hot set
            __builtin_nontemporal_store((_Float16)acc[nt][v], &C[(long)row * G4_ + col]);
        }
    }
}

// ---------------------------------------------------------------- scan init
__global__ void scan_init_kernel(_Float16* h0, _Float16* h1, int* cnt) {
    int i = blockIdx.x * blockDim.x + threadIdx.x;
    if (i < B_ * H_) { h0[i] = (_Float16)0.0f; h1[i] = (_Float16)0.0f; }
    if (i == 0) *cnt = 0;
}

// ---------------------------------------------------------------- LSTM scan
// Persistent cooperative kernel: NBLK_ blocks x 256 threads (8 waves).
// Block blk owns hidden cols [blk*32, blk*32+32) => 128 z-columns (4 gates).
// Wh slice lives in LDS (f16) for the whole scan; h(t-1) re-staged each step
// via GLOBAL_LOAD_ASYNC_TO_LDS_B128 (ASYNCcnt path, no VGPR round-trip).
__global__ void __launch_bounds__(256)
lstm_scan_kernel(const _Float16* __restrict__ XW,   // [B*T, 2048] f16 (x@Wi)
                 const float*    __restrict__ Wh,   // [512, 2048] f32
                 const float*    __restrict__ bias, // [2048]
                 _Float16* __restrict__ hbuf0,      // [64*512] f16 ping
                 _Float16* __restrict__ hbuf1,      // [64*512] f16 pong
                 _Float16* __restrict__ Hseq,       // [B,T,H] f16 or null
                 float*    __restrict__ hlast,      // [64*512] f32 or null
                 int*      __restrict__ bar) {
    extern __shared__ char smem_raw[];
    _Float16* whB = (_Float16*)smem_raw;            // [128][KPAD_]  (B operand, [n][k])
    _Float16* ha  = whB + ZPB_ * KPAD_;             // [64][KPAD_]   (A operand, [m][k])
    // byte offset of ha within this block's LDS allocation (no static LDS here)
    const unsigned ha_lds_base = (unsigned)((char*)ha - smem_raw);

    const int blk  = blockIdx.x;
    const int tid  = threadIdx.x;
    const int lane = tid & 31;
    const int wave = tid >> 5;
    const int mtile = wave >> 1;     // 0..3 (rows of batch)
    const int nhalf = wave & 1;      // 0..1 (16-col half of the 32-col slice)

    // One-time: stage this block's Wh columns into LDS as f16, [c][k] layout.
    for (int i = tid; i < ZPB_ * H_; i += 256) {
        int c = i & 127, k = i >> 7;
        int g = c >> 5, cc = c & 31;
        int col = g * H_ + blk * CPB_ + cc;
        whB[c * KPAD_ + k] = (_Float16)Wh[(long)k * G4_ + col];
    }
    __syncthreads();

    float cstate[8];
#pragma unroll
    for (int v = 0; v < 8; ++v) cstate[v] = 0.0f;

    float bgate[4];
    {
        int cc = nhalf * 16 + (lane & 15);
#pragma unroll
        for (int g = 0; g < 4; ++g) bgate[g] = bias[g * H_ + blk * CPB_ + cc];
    }

    for (int t = 0; t < T_; ++t) {
        const _Float16* hread  = (t & 1) ? hbuf1 : hbuf0;
        _Float16*       hwrite = (t & 1) ? hbuf0 : hbuf1;

        // stage h(t-1) [64 x 512] f16 into LDS: 4096 x 16B async copies
        // (row m: 1024B contiguous global -> 1024B contiguous LDS at m*KPAD_*2)
#pragma unroll 4
        for (int i = tid; i < B_ * 64; i += 256) {
            int m = i >> 6, c16 = i & 63;
            unsigned lds_off = ha_lds_base + (unsigned)(m * (KPAD_ * 2) + c16 * 16);
            uint64_t gaddr = (uint64_t)(uintptr_t)(hread + m * H_) + (uint64_t)(c16 * 16);
            asm volatile("global_load_async_to_lds_b128 %0, %1, off"
                         :: "v"(lds_off), "v"(gaddr) : "memory");
        }
        asm volatile("s_wait_asynccnt 0x0" ::: "memory");
        __syncthreads();

        v8f acc[4];
#pragma unroll
        for (int g = 0; g < 4; ++g) acc[g] = (v8f){0,0,0,0,0,0,0,0};

        // z[mtile, gate-tiles] += h @ Wh_slice    (K = 512 in 16 k-tiles)
        for (int kt = 0; kt < H_ / 32; ++kt) {
            v16h a = frag_a_lds(ha, KPAD_, mtile * 16, kt * 32, lane);
#pragma unroll
            for (int g = 0; g < 4; ++g) {
                v16h b = frag_b_lds(whB, KPAD_, (g * 2 + nhalf) * 16, kt * 32, lane);
                acc[g] = __builtin_amdgcn_wmma_f32_16x16x32_f16(
                    false, a, false, b, (short)0, acc[g], false, false);
            }
        }

        // gates: each lane owns 8 (m, hcol) elements, same coords in all 4 accs
        const int n = lane & 15, half = lane >> 4;
        const int cc = nhalf * 16 + n;
        const int hcol = blk * CPB_ + cc;
#pragma unroll
        for (int v = 0; v < 8; ++v) {
            int m = mtile * 16 + 8 * half + v;
            long rowoff = ((long)m * T_ + t) * G4_;
            // XW is streamed exactly once: non-temporal loads keep L2 for Wh/h
            float zi = acc[0][v] + (float)__builtin_nontemporal_load(XW + rowoff + 0 * H_ + hcol) + bgate[0];
            float zf = acc[1][v] + (float)__builtin_nontemporal_load(XW + rowoff + 1 * H_ + hcol) + bgate[1];
            float zg = acc[2][v] + (float)__builtin_nontemporal_load(XW + rowoff + 2 * H_ + hcol) + bgate[2];
            float zo = acc[3][v] + (float)__builtin_nontemporal_load(XW + rowoff + 3 * H_ + hcol) + bgate[3];
            float cn = sigf(zf) * cstate[v] + sigf(zi) * tanhfast(zg);
            cstate[v] = cn;
            float hn = sigf(zo) * tanhfast(cn);
            _Float16 hh = (_Float16)hn;
            hwrite[m * H_ + hcol] = hh;
            if (Hseq) __builtin_nontemporal_store(hh, &Hseq[((long)m * T_ + t) * H_ + hcol]);
            if (hlast && t == T_ - 1) hlast[m * H_ + hcol] = hn;
        }

        // ---- grid barrier (monotonic counter; device-scope release/acquire)
        __threadfence();
        __syncthreads();
        if (tid == 0) {
            __hip_atomic_fetch_add(bar, 1, __ATOMIC_RELEASE, __HIP_MEMORY_SCOPE_AGENT);
            const int target = (t + 1) * NBLK_;
            while (__hip_atomic_load(bar, __ATOMIC_ACQUIRE, __HIP_MEMORY_SCOPE_AGENT) < target)
                __builtin_amdgcn_s_sleep(2);
        }
        __syncthreads();
        __threadfence();
    }
}

// ---------------------------------------------------------------- dense head
__global__ void head0_kernel(const float* __restrict__ hlast, const float* __restrict__ x,
                             const float* __restrict__ Wd0, const float* __restrict__ bd0,
                             float* __restrict__ out1) {
    int i = blockIdx.x * blockDim.x + threadIdx.x;   // 64*512
    if (i >= B_ * 512) return;
    int m = i >> 9, nc = i & 511;
    float s = bd0[nc];
    for (int k = 0; k < H_; ++k) s += hlast[m * H_ + k] * Wd0[k * 512 + nc];
    for (int k = 0; k < 8; ++k)
        s += x[((long)m * T_ + (T_ - 1)) * FIN_ + FLSTM_ + k] * Wd0[(H_ + k) * 512 + nc];
    out1[i] = fmaxf(s, 0.0f);
}
__global__ void head1_kernel(const float* __restrict__ out1,
                             const float* __restrict__ Wd1, const float* __restrict__ bd1,
                             float* __restrict__ out2) {
    int i = blockIdx.x * blockDim.x + threadIdx.x;   // 64*256
    if (i >= B_ * 256) return;
    int m = i >> 8, nc = i & 255;
    float s = bd1[nc];
    for (int k = 0; k < 512; ++k) s += out1[m * 512 + k] * Wd1[k * 256 + nc];
    out2[i] = fmaxf(s, 0.0f);
}
__global__ void head2_kernel(const float* __restrict__ out2,
                             const float* __restrict__ Wf, const float* __restrict__ bf,
                             float* __restrict__ out) {
    int i = blockIdx.x * blockDim.x + threadIdx.x;   // 64*64
    if (i >= B_ * 64) return;
    int m = i >> 6, nc = i & 63;
    float s = bf[nc];
    for (int k = 0; k < 256; ++k) s += out2[m * 256 + k] * Wf[k * 64 + nc];
    out[i] = sigf(s);
}

// ---------------------------------------------------------------- launch
extern "C" void kernel_launch(void* const* d_in, const int* in_sizes, int n_in,
                              void* d_out, int out_size, void* d_ws, size_t ws_size,
                              hipStream_t stream) {
    (void)in_sizes; (void)n_in; (void)out_size; (void)ws_size;
    const float* x   = (const float*)d_in[0];
    const float* Wi0 = (const float*)d_in[1];
    const float* Wh0 = (const float*)d_in[2];
    const float* b0  = (const float*)d_in[3];
    const float* Wi1 = (const float*)d_in[4];
    const float* Wh1 = (const float*)d_in[5];
    const float* b1  = (const float*)d_in[6];
    const float* Wd0 = (const float*)d_in[7];
    const float* bd0 = (const float*)d_in[8];
    const float* Wd1 = (const float*)d_in[9];
    const float* bd1 = (const float*)d_in[10];
    const float* Wf  = (const float*)d_in[11];
    const float* bf  = (const float*)d_in[12];

    char* ws = (char*)d_ws;
    size_t off = 0;
    _Float16* XW  = (_Float16*)(ws + off); off += (size_t)B_ * T_ * G4_ * 2;  // 128 MB, reused by both layers
    _Float16* H1  = (_Float16*)(ws + off); off += (size_t)B_ * T_ * H_ * 2;   // 32 MB
    _Float16* hb0 = (_Float16*)(ws + off); off += (size_t)B_ * H_ * 2;
    _Float16* hb1 = (_Float16*)(ws + off); off += (size_t)B_ * H_ * 2;
    float* h2last = (float*)(ws + off);    off += (size_t)B_ * H_ * 4;
    float* out1   = (float*)(ws + off);    off += (size_t)B_ * 512 * 4;
    float* out2   = (float*)(ws + off);    off += (size_t)B_ * 256 * 4;
    int* bar      = (int*)(ws + off);      off += 256;

    (void)hipFuncSetAttribute((const void*)lstm_scan_kernel,
                              hipFuncAttributeMaxDynamicSharedMemorySize,
                              SCAN_LDS_BYTES);

    dim3 ggrid(16, 512);   // 2048/128 cols x 32768/64 rows

    // ---- layer 0
    gemm_xw_kernel<false><<<ggrid, 256, 0, stream>>>((const void*)x, FIN_, FLSTM_, Wi0, XW);
    scan_init_kernel<<<(B_ * H_ + 255) / 256, 256, 0, stream>>>(hb0, hb1, bar);
    lstm_scan_kernel<<<NBLK_, 256, SCAN_LDS_BYTES, stream>>>(XW, Wh0, b0, hb0, hb1,
                                                             H1, nullptr, bar);
    // ---- layer 1
    gemm_xw_kernel<true><<<ggrid, 256, 0, stream>>>((const void*)H1, H_, H_, Wi1, XW);
    scan_init_kernel<<<(B_ * H_ + 255) / 256, 256, 0, stream>>>(hb0, hb1, bar);
    lstm_scan_kernel<<<NBLK_, 256, SCAN_LDS_BYTES, stream>>>(XW, Wh1, b1, hb0, hb1,
                                                             nullptr, h2last, bar);
    // ---- head
    head0_kernel<<<(B_ * 512 + 255) / 256, 256, 0, stream>>>(h2last, x, Wd0, bd0, out1);
    head1_kernel<<<(B_ * 256 + 255) / 256, 256, 0, stream>>>(out1, Wd1, bd1, out2);
    head2_kernel<<<(B_ * 64 + 255) / 256, 256, 0, stream>>>(out2, Wf, bf, (float*)d_out);
}
